// RotationV3_2164663517290
// MI455X (gfx1250) — matrix operand order repaired and checked by reading, not compiled, not run on hardware
//
#include <hip/hip_runtime.h>
#include <hip/hip_bf16.h>

#define S_  1024
#define D_  1024
#define P_  64
#define C_  28
#define V_  151936
#define EPS_ 1.1920929e-07f

typedef __attribute__((ext_vector_type(16))) __bf16 v16bf;
typedef __attribute__((ext_vector_type(8)))  float  v8f;

// exact parameter types of __builtin_amdgcn_global_load_async_to_lds_b128
typedef __attribute__((__vector_size__(16))) int v4i;
typedef __attribute__((address_space(1))) v4i* gptr_v4i;
typedef __attribute__((address_space(3))) v4i* lptr_v4i;

union BF16x16 {
    v16bf v;
    unsigned short u[16];
    unsigned d[8];
    uint4 q[2];
};

// round-to-nearest-even f32 -> bf16 (scalar fallback)
static __device__ __forceinline__ unsigned short f2bf(float f) {
    unsigned u = __float_as_uint(f);
    u += 0x7FFFu + ((u >> 16) & 1u);
    return (unsigned short)(u >> 16);
}

// packed f32x2 -> bf16x2 (one v_cvt_pk_bf16_f32 when available)
static __device__ __forceinline__ unsigned cvt2bf(float lo, float hi) {
#if __has_builtin(__builtin_amdgcn_cvt_pk_bf16_f32)
    typedef __attribute__((ext_vector_type(2))) __bf16 v2bf;
    union { v2bf v; unsigned u; } t;
    t.v = __builtin_amdgcn_cvt_pk_bf16_f32(lo, hi);
    return t.u;
#else
    return (unsigned)f2bf(lo) | ((unsigned)f2bf(hi) << 16);
#endif
}

// ---- CDNA5 async copy: global -> LDS, 16 bytes per lane, tracked by ASYNCcnt
static __device__ __forceinline__ void cp_async_b128(void* lds_generic, const void* gsrc) {
#if __has_builtin(__builtin_amdgcn_global_load_async_to_lds_b128)
    void* g = const_cast<void*>(gsrc);
    __builtin_amdgcn_global_load_async_to_lds_b128(
        (gptr_v4i)g, (lptr_v4i)lds_generic, 0 /*imm offset*/, 0 /*cpol*/);
#else
    *(uint4*)lds_generic = *(const uint4*)gsrc;   // synchronous fallback
#endif
}

static __device__ __forceinline__ void wait_async0() {
#if __has_builtin(__builtin_amdgcn_global_load_async_to_lds_b128)
#if __has_builtin(__builtin_amdgcn_s_wait_asynccnt)
    __builtin_amdgcn_s_wait_asynccnt(0);
#else
    asm volatile("s_wait_asynccnt 0x0" ::: "memory");
#endif
#endif
}

// ---------------------------------------------------------------- embed gather
__global__ __launch_bounds__(256) void k_embed(const int* __restrict__ tok,
                                               const float* __restrict__ tbl,
                                               float* __restrict__ x) {
    int s = blockIdx.x;
    size_t row = (size_t)tok[s] * D_;
    for (int j = 0; j < 4; ++j) {
        int d = threadIdx.x + j * 256;
        x[(size_t)s * D_ + d] = tbl[row + d];
    }
}

// ---------------------------------------------------------------- column mean over S (8-way split, atomic)
__global__ __launch_bounds__(256) void k_colmean(const float* __restrict__ x,
                                                 float* __restrict__ meanD) {
    int d  = blockIdx.x * 256 + threadIdx.x;
    int t0 = blockIdx.y * (S_ / 8);
    float acc = 0.f;
    for (int t = t0; t < t0 + S_ / 8; ++t) acc += x[(size_t)t * D_ + d];
    atomicAdd(&meanD[d], acc * (1.0f / (float)S_));
}

// ---------------------------------------------------------------- angles -> cos/sin
__global__ __launch_bounds__(256) void k_angles(const float* __restrict__ meanD,
                                                const float* __restrict__ agw,
                                                const float* __restrict__ bang,
                                                float* __restrict__ ca,
                                                float* __restrict__ sa) {
    __shared__ float red[256];
    int p = blockIdx.x, t = threadIdx.x;
    float acc = 0.f;
    for (int j = 0; j < 4; ++j) {
        int d = t + j * 256;
        acc += meanD[d] * agw[(size_t)p * D_ + d];
    }
    red[t] = acc; __syncthreads();
    for (int o = 128; o > 0; o >>= 1) {
        if (t < o) red[t] += red[t + o];
        __syncthreads();
    }
    if (t == 0) {
        float ang = bang[p] + 0.1f * red[0];
        ca[p] = __cosf(ang);
        sa[p] = __sinf(ang);
    }
}

// ---------------------------------------------------------------- EMA causal scan, bf16 out
__global__ __launch_bounds__(256) void k_ema(const float* __restrict__ x,
                                             const float* __restrict__ decay,
                                             unsigned short* __restrict__ cb) {
    int d = blockIdx.x * 256 + threadIdx.x;
    float dv = 1.0f / (1.0f + __expf(-decay[0]));
    float od = 1.0f - dv;
    float s = 0.f;
    for (int t = 0; t < S_; ++t) {
        s = dv * s + od * x[(size_t)t * D_ + d];
        cb[(size_t)t * D_ + d] = f2bf(s);
    }
}

// ---------------------------------------------------------------- f32 -> bf16 bulk convert (8 elems/thread)
__global__ __launch_bounds__(256) void k_cvt(const float* __restrict__ src,
                                             unsigned short* __restrict__ dst) {
    size_t base = ((size_t)blockIdx.x * 256 + threadIdx.x) * 8;
    float4 f0 = *(const float4*)(src + base);
    float4 f1 = *(const float4*)(src + base + 4);
    uint4 o;
    o.x = cvt2bf(f0.x, f0.y);
    o.y = cvt2bf(f0.z, f0.w);
    o.z = cvt2bf(f1.x, f1.y);
    o.w = cvt2bf(f1.z, f1.w);
    *(uint4*)(dst + base) = o;
}

// ---------------------------------------------------------------- plane rotation, bf16 out
__global__ __launch_bounds__(128) void k_rot(const float* __restrict__ proj,
                                             const float* __restrict__ ca,
                                             const float* __restrict__ sa,
                                             unsigned short* __restrict__ rotb) {
    int s = blockIdx.x, t = threadIdx.x;
    const float* pr = proj + (size_t)s * 128;
    float v;
    if (t < 64) v = pr[t] * ca[t] - pr[t + 64] * sa[t];
    else { int q = t - 64; v = pr[q] * sa[q] + pr[q + 64] * ca[q]; }
    rotb[(size_t)s * 128 + t] = f2bf(v);
}

// ---------------------------------------------------------------- fused SiLU + rms residual update
__global__ __launch_bounds__(256) void k_post(const float* __restrict__ xrot,
                                              float* __restrict__ x,
                                              const float* __restrict__ sc,
                                              const float* __restrict__ sh,
                                              const float* __restrict__ nw) {
    __shared__ float red[256];
    int s = blockIdx.x, t = threadIdx.x;
    float xv[4], vv[4];
    float acc = 0.f;
    for (int j = 0; j < 4; ++j) {
        int d = t + j * 256;
        size_t idx = (size_t)s * D_ + d;
        float xi = x[idx];
        float z = xrot[idx] * sc[d] + sh[d];
        float act = z / (1.0f + __expf(-z));
        xv[j] = xi;
        vv[j] = act - xi;
        acc += vv[j] * vv[j];
    }
    red[t] = acc; __syncthreads();
    for (int o = 128; o > 0; o >>= 1) {
        if (t < o) red[t] += red[t + o];
        __syncthreads();
    }
    float r = rsqrtf(red[0] * (1.0f / (float)D_) + EPS_);
    for (int j = 0; j < 4; ++j) {
        int d = t + j * 256;
        x[(size_t)s * D_ + d] = xv[j] + vv[j] * r * nw[d];
    }
}

// ---------------------------------------------------------------- out rms-norm -> bf16
__global__ __launch_bounds__(256) void k_outnorm(const float* __restrict__ x,
                                                 const float* __restrict__ onw,
                                                 unsigned short* __restrict__ xa) {
    __shared__ float red[256];
    int s = blockIdx.x, t = threadIdx.x;
    float xv[4];
    float acc = 0.f;
    for (int j = 0; j < 4; ++j) {
        int d = t + j * 256;
        xv[j] = x[(size_t)s * D_ + d];
        acc += xv[j] * xv[j];
    }
    red[t] = acc; __syncthreads();
    for (int o = 128; o > 0; o >>= 1) {
        if (t < o) red[t] += red[t + o];
        __syncthreads();
    }
    float r = rsqrtf(red[0] * (1.0f / (float)D_) + EPS_);
    for (int j = 0; j < 4; ++j) {
        int d = t + j * 256;
        xa[(size_t)s * D_ + d] = f2bf(xv[j] * r * onw[d]);
    }
}

// ---------------------------------------------------------------- pure-bf16 WMMA GEMM, async-LDS staged W
// OUT[M,N] = (addres ? RES : 0) + A[M,K](bf16) @ W[N,K](bf16)^T
// block = 256 threads = 8 waves; tile 128(M) x 128(N); K tile 64 (2 WMMA half-steps).
// W tile 128 x 64 bf16, LDS rows padded to 72 elems (conflict-free ds_load_b128).
#define WROW_ 72
__global__ __launch_bounds__(256) void k_gemm_bf16(const unsigned short* __restrict__ A,
                                                   const unsigned short* __restrict__ Wb,
                                                   const float* __restrict__ RES,
                                                   float* __restrict__ OUT,
                                                   int M, int N, int K, int addres) {
    __shared__ alignas(16) unsigned short wt[2][128 * WROW_];

    int lane = threadIdx.x & 31;
    int wave = threadIdx.x >> 5;
    int wm = wave & 3, wn = wave >> 2;
    int m0 = blockIdx.y * 128 + wm * 32;
    int nblk = blockIdx.x * 128;
    int n0 = nblk + wn * 64;
    int r = lane & 15, h = lane >> 4;

    v8f acc[2][4];
    for (int i = 0; i < 2; ++i)
        for (int j = 0; j < 4; ++j)
            acc[i][j] = (v8f)(0.0f);

    // async copy of one 128x64 bf16 W tile into wt[buf]
    auto load_tile = [&](int buf, int k0) {
        int t = threadIdx.x;
        for (int i = 0; i < 4; ++i) {
            int c   = t + i * 256;        // 0..1023 16B chunks
            int row = c >> 3;             // 0..127
            int kq  = (c & 7) * 8;        // bf16 elem offset 0..56
            cp_async_b128(&wt[buf][row * WROW_ + kq],
                          Wb + (size_t)(nblk + row) * K + (k0 + kq));
        }
    };

    int buf = 0;
    load_tile(0, 0);
    wait_async0();
    __syncthreads();

    for (int k0 = 0; k0 < K; k0 += 64) {
        bool more = (k0 + 64) < K;
        if (more) load_tile(buf ^ 1, k0 + 64);

        for (int ks2 = 0; ks2 < 2; ++ks2) {
            int ks = ks2 * 32;
            // A fragments direct from global (bf16, K-contiguous)
            BF16x16 af[2];
            for (int i = 0; i < 2; ++i) {
                const unsigned short* ap =
                    A + (size_t)(m0 + i * 16 + r) * K + (k0 + ks + h * 8);
                af[i].q[0] = *(const uint4*)ap;
                af[i].q[1] = *(const uint4*)(ap + 16);
            }
            // B fragments: two ds_load_b128, zero conversion
            BF16x16 bf[4];
            for (int j = 0; j < 4; ++j) {
                const unsigned short* wrow =
                    &wt[buf][(wn * 64 + j * 16 + r) * WROW_ + ks + h * 8];
                bf[j].q[0] = *(const uint4*)wrow;
                bf[j].q[1] = *(const uint4*)(wrow + 16);
            }
            for (int i = 0; i < 2; ++i)
                for (int j = 0; j < 4; ++j)
                    acc[i][j] = __builtin_amdgcn_wmma_f32_16x16x32_bf16(
                        false, af[i].v, false, bf[j].v, (short)0, acc[i][j], false, false);
        }

        if (more) wait_async0();
        __syncthreads();
        buf ^= 1;
    }

    if (addres) {
        for (int i = 0; i < 2; ++i)
            for (int j = 0; j < 4; ++j)
                for (int v = 0; v < 8; ++v) {
                    int row = m0 + i * 16 + h * 8 + v;
                    int col = n0 + j * 16 + (lane & 15);
                    size_t idx = (size_t)row * N + col;
                    OUT[idx] = acc[i][j][v] + RES[idx];
                }
    } else {
        for (int i = 0; i < 2; ++i)
            for (int j = 0; j < 4; ++j)
                for (int v = 0; v < 8; ++v) {
                    int row = m0 + i * 16 + h * 8 + v;
                    int col = n0 + j * 16 + (lane & 15);
                    OUT[(size_t)row * N + col] = acc[i][j][v];
                }
    }
}

// ---------------------------------------------------------------- fallback GEMM: f32 W, in-register cvt
__global__ __launch_bounds__(256) void k_gemm_f32w(const unsigned short* __restrict__ A,
                                                   const float* __restrict__ W,
                                                   float* __restrict__ OUT,
                                                   int M, int N, int K) {
    __shared__ alignas(16) float wtile[2][128 * 32];

    int lane = threadIdx.x & 31;
    int wave = threadIdx.x >> 5;
    int wm = wave & 3, wn = wave >> 2;
    int m0 = blockIdx.y * 128 + wm * 32;
    int nblk = blockIdx.x * 128;
    int n0 = nblk + wn * 64;
    int r = lane & 15, h = lane >> 4;

    v8f acc[2][4];
    for (int i = 0; i < 2; ++i)
        for (int j = 0; j < 4; ++j)
            acc[i][j] = (v8f)(0.0f);

    auto load_tile = [&](int buf, int k0) {
        int t = threadIdx.x;
        for (int i = 0; i < 4; ++i) {
            int c   = t + i * 256;
            int row = c >> 3;
            int kq  = (c & 7) * 4;
            cp_async_b128(&wtile[buf][row * 32 + kq],
                          W + (size_t)(nblk + row) * K + (k0 + kq));
        }
    };

    int buf = 0;
    load_tile(0, 0);
    wait_async0();
    __syncthreads();

    for (int k0 = 0; k0 < K; k0 += 32) {
        bool more = (k0 + 32) < K;
        if (more) load_tile(buf ^ 1, k0 + 32);

        BF16x16 af[2];
        for (int i = 0; i < 2; ++i) {
            const unsigned short* ap =
                A + (size_t)(m0 + i * 16 + r) * K + (k0 + h * 16);
            af[i].q[0] = *(const uint4*)ap;
            af[i].q[1] = *(const uint4*)(ap + 8);
        }
        BF16x16 bf[4];
        for (int j = 0; j < 4; ++j) {
            const float4* wp =
                (const float4*)&wtile[buf][(wn * 64 + j * 16 + r) * 32 + h * 16];
            for (int c = 0; c < 4; ++c) {
                float4 f = wp[c];
                bf[j].d[2 * c + 0] = cvt2bf(f.x, f.y);
                bf[j].d[2 * c + 1] = cvt2bf(f.z, f.w);
            }
        }
        for (int i = 0; i < 2; ++i)
            for (int j = 0; j < 4; ++j)
                acc[i][j] = __builtin_amdgcn_wmma_f32_16x16x32_bf16(
                    false, af[i].v, false, bf[j].v, (short)0, acc[i][j], false, false);

        if (more) wait_async0();
        __syncthreads();
        buf ^= 1;
    }

    for (int i = 0; i < 2; ++i)
        for (int j = 0; j < 4; ++j)
            for (int v = 0; v < 8; ++v) {
                int row = m0 + i * 16 + h * 8 + v;
                int col = n0 + j * 16 + (lane & 15);
                OUT[(size_t)row * N + col] = acc[i][j][v];
            }
}

// ---------------------------------------------------------------- host orchestration
extern "C" void kernel_launch(void* const* d_in, const int* in_sizes, int n_in,
                              void* d_out, int out_size, void* d_ws, size_t ws_size,
                              hipStream_t stream) {
    const int*   tokens = (const int*)d_in[0];
    const float* embed  = (const float*)d_in[1];
    const float* ppw    = (const float*)d_in[2];   // (C, 2P, D)
    const float* bpw    = (const float*)d_in[3];   // (C, D, 2P)
    const float* bang   = (const float*)d_in[4];   // (C, P)
    const float* agw    = (const float*)d_in[5];   // (P, D)
    const float* scales = (const float*)d_in[6];   // (C, D)
    const float* shifts = (const float*)d_in[7];   // (C, D)
    const float* normw  = (const float*)d_in[8];   // (C, D)
    const float* decay  = (const float*)d_in[9];   // scalar
    const float* gatew  = (const float*)d_in[10];  // (D, D)
    const float* onw    = (const float*)d_in[11];  // (D,)
    const float* lmw    = (const float*)d_in[12];  // (V, D)

    char* p = (char*)d_ws;
    auto alloc = [&](size_t bytes) {
        char* r = p;
        p += (bytes + 255) & ~(size_t)255;
        return r;
    };
    float*          x      = (float*)alloc((size_t)S_ * D_ * 4);
    float*          xrot   = (float*)alloc((size_t)S_ * D_ * 4);
    float*          proj   = (float*)alloc((size_t)S_ * 128 * 4);
    float*          meanD  = (float*)alloc(D_ * 4);
    float*          ca     = (float*)alloc(P_ * 4);
    float*          sa     = (float*)alloc(P_ * 4);
    unsigned short* xa     = (unsigned short*)alloc((size_t)S_ * D_ * 2);
    unsigned short* rotb   = (unsigned short*)alloc((size_t)S_ * 128 * 2);
    unsigned short* cb     = (unsigned short*)alloc((size_t)S_ * D_ * 2);
    unsigned short* ppw_b  = (unsigned short*)alloc((size_t)C_ * 128 * D_ * 2);
    unsigned short* bpw_b  = (unsigned short*)alloc((size_t)C_ * D_ * 128 * 2);
    unsigned short* gate_b = (unsigned short*)alloc((size_t)D_ * D_ * 2);

    size_t lmw_bytes = (size_t)V_ * D_ * 2;
    size_t used = (size_t)(p - (char*)d_ws);
    unsigned short* lmw_b =
        (used + lmw_bytes + 256 <= ws_size) ? (unsigned short*)alloc(lmw_bytes) : nullptr;

    auto gemm = [&](const unsigned short* A, const unsigned short* Wb, const float* RES,
                    float* OUT, int M, int N, int K, int addres) {
        dim3 g(N / 128, M / 128);
        k_gemm_bf16<<<g, 256, 0, stream>>>(A, Wb, RES, OUT, M, N, K, addres);
    };

    // pre-convert constant weights to bf16 (once per launch)
    k_cvt<<<((size_t)C_ * 128 * D_) / 2048, 256, 0, stream>>>(ppw, ppw_b);
    k_cvt<<<((size_t)C_ * D_ * 128) / 2048, 256, 0, stream>>>(bpw, bpw_b);
    k_cvt<<<((size_t)D_ * D_) / 2048, 256, 0, stream>>>(gatew, gate_b);
    if (lmw_b)
        k_cvt<<<((size_t)V_ * D_) / 2048, 256, 0, stream>>>(lmw, lmw_b);

    k_embed<<<S_, 256, 0, stream>>>(tokens, embed, x);

    for (int i = 0; i < C_; ++i) {
        (void)hipMemsetAsync(meanD, 0, D_ * sizeof(float), stream);
        k_colmean<<<dim3(D_ / 256, 8), 256, 0, stream>>>(x, meanD);
        k_angles<<<P_, 256, 0, stream>>>(meanD, agw, bang + (size_t)i * P_, ca, sa);
        if ((i & 3) == 0) {
            k_ema<<<D_ / 256, 256, 0, stream>>>(x, decay, cb);
            gemm(cb, gate_b, x, x, S_, D_, D_, 1);           // x += causal @ gate^T
        }
        k_cvt<<<(S_ * D_) / 2048, 256, 0, stream>>>(x, xa);
        gemm(xa, ppw_b + (size_t)i * 128 * D_, nullptr, proj, S_, 128, D_, 0);
        k_rot<<<S_, 128, 0, stream>>>(proj, ca, sa, rotb);
        gemm(rotb, bpw_b + (size_t)i * D_ * 128, x, xrot, S_, D_, 128, 1);
        k_post<<<S_, 256, 0, stream>>>(xrot, x,
                                       scales + (size_t)i * D_,
                                       shifts + (size_t)i * D_,
                                       normw + (size_t)i * D_);
    }

    k_outnorm<<<S_, 256, 0, stream>>>(x, onw, xa);
    if (lmw_b) {
        gemm(xa, lmw_b, nullptr, (float*)d_out, S_, V_, D_, 0);
    } else {
        dim3 g(V_ / 128, S_ / 128);
        k_gemm_f32w<<<g, 256, 0, stream>>>(xa, lmw, (float*)d_out, S_, V_, D_);
    }
}
